// Llama4TextExpertsWithLoRA_72267119722648
// MI455X (gfx1250) — matrix operand order repaired and checked by reading, not compiled, not run on hardware
//
#include <hip/hip_runtime.h>
#include <cstdint>

// Problem constants (from reference)
#define E_  8
#define H_  1024
#define I_  2048
#define F2_ 4096
#define T_  1024
#define R_  8

typedef __attribute__((ext_vector_type(16))) __bf16          v16bf;
typedef __attribute__((ext_vector_type(8)))  float           v8f;
typedef __attribute__((ext_vector_type(4)))  int             v4i;
typedef __attribute__((ext_vector_type(8)))  unsigned short  ushort8;
typedef __attribute__((ext_vector_type(16))) unsigned short  ushort16;

// Optional gfx1250 async global->LDS path (guarded; falls back to VGPR copy)
#if defined(__AMDGCN__) && defined(__has_builtin)
#if __has_builtin(__builtin_amdgcn_global_load_async_to_lds_b128)
#define HAVE_ASYNC_LDS 1
#endif
#if __has_builtin(__builtin_amdgcn_s_wait_asynccnt)
#define HAVE_WAIT_ASYNC 1
#endif
#endif

#ifdef HAVE_ASYNC_LDS
__device__ __forceinline__ void async_copy_b128(const void* gsrc, void* ldst) {
  // Signature (from hipcc diagnostic): pointers to 128-bit int vectors.
  __builtin_amdgcn_global_load_async_to_lds_b128(
      (v4i*)gsrc, (v4i*)ldst, 0, 0);
}
__device__ __forceinline__ void wait_async0() {
#ifdef HAVE_WAIT_ASYNC
  __builtin_amdgcn_s_wait_asynccnt(0);
#else
  asm volatile("s_wait_asynccnt 0" ::: "memory");
#endif
}
#endif

// fp32 -> bf16 (round to nearest even, NaN ignored for this data)
__device__ __forceinline__ unsigned short f2bf(float f) {
  unsigned int u = __builtin_bit_cast(unsigned int, f);
  u += 0x7FFFu + ((u >> 16) & 1u);
  return (unsigned short)(u >> 16);
}
__device__ __forceinline__ float bf2f(unsigned short u) {
  return __builtin_bit_cast(float, ((unsigned int)u) << 16);
}

__device__ __forceinline__ v8f wmma_bf16(v16bf a, v16bf b, v8f c) {
  // D = A(16x32 bf16) x B(32x16 bf16) + C(16x16 f32)
  return __builtin_amdgcn_wmma_f32_16x16x32_bf16(false, a, false, b, (short)0, c,
                                                 false, false);
}

#define LDK 40  // LDS row pitch in ushort (80B = 5x16B, keeps 16B alignment)

// A fragment per ISA: lanes 0-15 hold row m, K = {0..7, 16..23};
// lanes 16-31 hold row m, K = {8..15, 24..31}.
__device__ __forceinline__ v16bf frag_a(const unsigned short* base, int m, int lane) {
  const int kh = (lane & 16) ? 8 : 0;
  const unsigned short* p = base + m * LDK;
  ushort8 lo = *(const ushort8*)(p + kh);
  ushort8 hi = *(const ushort8*)(p + 16 + kh);
  ushort16 v = __builtin_shufflevector(lo, hi, 0, 1, 2, 3, 4, 5, 6, 7,
                                               8, 9, 10, 11, 12, 13, 14, 15);
  return __builtin_bit_cast(v16bf, v);
}
// B fragment per ISA: lanes 0-15 hold column n with K = {0..7, 16..23},
// lanes 16-31 hold column n with K = {8..15, 24..31}. LDS tile stored [N][K] (K-contiguous).
__device__ __forceinline__ v16bf frag_b(const unsigned short* base, int col, int lane) {
  const int ks = (lane & 16) ? 8 : 0;
  const unsigned short* p = base + col * LDK + ks;
  ushort8 lo = *(const ushort8*)(p);
  ushort8 hi = *(const ushort8*)(p + 16);
  ushort16 v = __builtin_shufflevector(lo, hi, 0, 1, 2, 3, 4, 5, 6, 7,
                                               8, 9, 10, 11, 12, 13, 14, 15);
  return __builtin_bit_cast(v16bf, v);
}

// ---------------------------------------------------------------------------
// s = 2.0 * (x @ A^T), x fp32 [8192, H_], A fp32 [R_, H_] -> s [8192, R_]
// one wave per row; lane-strided K, shfl-xor reduction.
// ---------------------------------------------------------------------------
__global__ __launch_bounds__(256) void lora_s_gu_kernel(
    const float* __restrict__ x, const float* __restrict__ A,
    float* __restrict__ s) {
  const int lane = threadIdx.x & 31;
  const int row  = blockIdx.x * 8 + (threadIdx.x >> 5);
  const float* xp = x + (size_t)row * H_;
  float acc[R_];
#pragma unroll
  for (int r = 0; r < R_; ++r) acc[r] = 0.f;
  for (int k = lane; k < H_; k += 32) {
    const float xv = xp[k];
#pragma unroll
    for (int r = 0; r < R_; ++r) acc[r] += xv * A[r * H_ + k];
  }
#pragma unroll
  for (int r = 0; r < R_; ++r)
#pragma unroll
    for (int off = 16; off > 0; off >>= 1) acc[r] += __shfl_xor(acc[r], off, 32);
  if (lane == 0) {
#pragma unroll
    for (int r = 0; r < R_; ++r) s[(size_t)row * R_ + r] = 2.0f * acc[r];
  }
}

// Same but A is over INTER dim and x is the bf16 hidden in workspace.
__global__ __launch_bounds__(256) void lora_s_dn_kernel(
    const unsigned short* __restrict__ hid, const float* __restrict__ A,
    float* __restrict__ s) {
  const int lane = threadIdx.x & 31;
  const int row  = blockIdx.x * 8 + (threadIdx.x >> 5);
  const unsigned short* hp = hid + (size_t)row * I_;
  float acc[R_];
#pragma unroll
  for (int r = 0; r < R_; ++r) acc[r] = 0.f;
  for (int k = lane; k < I_; k += 32) {
    const float xv = bf2f(hp[k]);
#pragma unroll
    for (int r = 0; r < R_; ++r) acc[r] += xv * A[r * I_ + k];
  }
#pragma unroll
  for (int r = 0; r < R_; ++r)
#pragma unroll
    for (int off = 16; off > 0; off >>= 1) acc[r] += __shfl_xor(acc[r], off, 32);
  if (lane == 0) {
#pragma unroll
    for (int r = 0; r < R_; ++r) s[(size_t)row * R_ + r] = 2.0f * acc[r];
  }
}

// ---------------------------------------------------------------------------
// GEMM1 + LoRA + SwiGLU fused. Block tile: 128 rows x 64 intermediate cols
// (gate cols [i0, i0+64) and up cols [i0+I_, ...)). 8 waves, each owns one
// 16-row stripe and 4 gate + 4 up 16x16 accumulator tiles.
// Grid: blockIdx.x = M index (fastest) so blocks sharing a weight strip are
// launched back-to-back and hit the 192MB L2 -> weights stream from HBM once.
// ---------------------------------------------------------------------------
__global__ __launch_bounds__(256) void gemm_gu_swiglu_kernel(
    const float* __restrict__ x,    // [E_*T_, H_]
    const float* __restrict__ w,    // [E_, H_, F2_]
    const float* __restrict__ Bgu,  // [F2_, R_]
    const float* __restrict__ sgu,  // [E_*T_, R_] (pre-scaled by 2.0)
    unsigned short* __restrict__ hidden) {  // [E_*T_, I_] bf16
  __shared__ __align__(16) unsigned short As[128 * LDK];
  __shared__ __align__(16) unsigned short Wg[64 * LDK];
  __shared__ __align__(16) unsigned short Wu[64 * LDK];

  const int tid = threadIdx.x;
  const int lane = tid & 31, wave = tid >> 5;
  const int e  = blockIdx.z;
  const int m0 = blockIdx.x * 128;   // M fastest-varying
  const int i0 = blockIdx.y * 64;

  const float* xe = x + ((size_t)e * T_ + m0) * H_;
  const float* we = w + (size_t)e * H_ * F2_;

  const v8f vzero = {0.f, 0.f, 0.f, 0.f, 0.f, 0.f, 0.f, 0.f};
  v8f accg[4], accu[4];
#pragma unroll
  for (int n = 0; n < 4; ++n) { accg[n] = vzero; accu[n] = vzero; }

  const int xr = tid >> 3;       // 0..31
  const int xc = (tid & 7) * 4;  // 0..28
  const int wkr = tid >> 4;      // 0..15
  const int wf  = (tid & 15) * 4;// 0..60

  for (int kk = 0; kk < H_; kk += 32) {
    __syncthreads();
    // x tile: 128x32 fp32 -> bf16 LDS
#pragma unroll
    for (int rr = 0; rr < 4; ++rr) {
      const int row = xr + rr * 32;
      const float4 v = *(const float4*)(xe + (size_t)row * H_ + kk + xc);
      unsigned short* d = &As[row * LDK + xc];
      d[0] = f2bf(v.x); d[1] = f2bf(v.y); d[2] = f2bf(v.z); d[3] = f2bf(v.w);
    }
    // weight tiles: 32x64 fp32, transposed into LDS as [N][K] bf16
#pragma unroll
    for (int kk2 = 0; kk2 < 2; ++kk2) {
      const int k = wkr + kk2 * 16;
      const float* wrow = we + (size_t)(kk + k) * F2_;
      const float4 g = *(const float4*)(wrow + i0 + wf);
      const float4 u = *(const float4*)(wrow + I_ + i0 + wf);
      Wg[(wf + 0) * LDK + k] = f2bf(g.x); Wg[(wf + 1) * LDK + k] = f2bf(g.y);
      Wg[(wf + 2) * LDK + k] = f2bf(g.z); Wg[(wf + 3) * LDK + k] = f2bf(g.w);
      Wu[(wf + 0) * LDK + k] = f2bf(u.x); Wu[(wf + 1) * LDK + k] = f2bf(u.y);
      Wu[(wf + 2) * LDK + k] = f2bf(u.z); Wu[(wf + 3) * LDK + k] = f2bf(u.w);
    }
    __syncthreads();

    const v16bf a = frag_a(As, wave * 16 + (lane & 15), lane);
#pragma unroll
    for (int nt = 0; nt < 4; ++nt) {
      const v16bf bg = frag_b(Wg, nt * 16 + (lane & 15), lane);
      accg[nt] = wmma_bf16(a, bg, accg[nt]);
      const v16bf bu = frag_b(Wu, nt * 16 + (lane & 15), lane);
      accu[nt] = wmma_bf16(a, bu, accu[nt]);
    }
  }

  // Epilogue: + rank-8 LoRA, SwiGLU, store bf16 hidden.
  // C/D layout: VGPR v, lanes 0-15 -> (M=v, N=lane); lanes 16-31 -> (M=v+8, N=lane-16)
  const int hi8 = (lane & 16) ? 8 : 0;
#pragma unroll
  for (int nt = 0; nt < 4; ++nt) {
    const int iCol = i0 + nt * 16 + (lane & 15);
    float bgr[R_], bur[R_];
#pragma unroll
    for (int r = 0; r < R_; ++r) {
      bgr[r] = Bgu[(size_t)iCol * R_ + r];
      bur[r] = Bgu[(size_t)(iCol + I_) * R_ + r];
    }
#pragma unroll
    for (int v = 0; v < 8; ++v) {
      const int t = m0 + wave * 16 + v + hi8;
      const size_t row = (size_t)e * T_ + t;
      const float* sp = sgu + row * R_;
      float g = accg[nt][v], u = accu[nt][v];
#pragma unroll
      for (int r = 0; r < R_; ++r) { g += sp[r] * bgr[r]; u += sp[r] * bur[r]; }
      const float hsw = u * g / (1.f + __expf(-g));  // up * silu(gate)
      hidden[row * I_ + iCol] = f2bf(hsw);
    }
  }
}

// ---------------------------------------------------------------------------
// GEMM2 + LoRA. Block tile 128 rows x 128 out cols; A is bf16 hidden from ws
// (copied via ASYNC global->LDS when the builtin is available).
// ---------------------------------------------------------------------------
__global__ __launch_bounds__(256) void gemm_down_kernel(
    const unsigned short* __restrict__ hidden,  // [E_*T_, I_] bf16
    const float* __restrict__ w,                // [E_, I_, H_]
    const float* __restrict__ Bdn,              // [H_, R_]
    const float* __restrict__ sdn,              // [E_*T_, R_] (pre-scaled)
    float* __restrict__ out) {                  // [E_*T_, H_]
  __shared__ __align__(16) unsigned short As[128 * LDK];
  __shared__ __align__(16) unsigned short Wd[128 * LDK];

  const int tid = threadIdx.x;
  const int lane = tid & 31, wave = tid >> 5;
  const int e  = blockIdx.z;
  const int m0 = blockIdx.x * 128;   // M fastest-varying (share W strip in L2)
  const int h0 = blockIdx.y * 128;

  const unsigned short* he = hidden + ((size_t)e * T_ + m0) * I_;
  const float* we = w + (size_t)e * I_ * H_;

  const v8f vzero = {0.f, 0.f, 0.f, 0.f, 0.f, 0.f, 0.f, 0.f};
  v8f acc[8];
#pragma unroll
  for (int n = 0; n < 8; ++n) acc[n] = vzero;

  const int ar = tid >> 1;        // 0..127
  const int ac = (tid & 1) * 16;  // 0 or 16
  const int wkr = tid >> 5;       // 0..7
  const int wf  = (tid & 31) * 4; // 0..124

  for (int kk = 0; kk < I_; kk += 32) {
    __syncthreads();
    // A tile: 128x32 bf16, straight copy (async -> LDS on CDNA5 toolchains)
    {
      const unsigned short* gp = he + (size_t)ar * I_ + kk + ac;
      unsigned short* lp = &As[ar * LDK + ac];
#ifdef HAVE_ASYNC_LDS
      async_copy_b128(gp, lp);
      async_copy_b128(gp + 8, lp + 8);
#else
      const ushort8* p = (const ushort8*)gp;
      *(ushort8*)lp = p[0];
      *(ushort8*)(lp + 8) = p[1];
#endif
    }
#pragma unroll
    for (int kk2 = 0; kk2 < 4; ++kk2) {  // W tile: 32x128 fp32 -> [N][K] bf16
      const int k = wkr + kk2 * 8;
      const float4 v = *(const float4*)(we + (size_t)(kk + k) * H_ + h0 + wf);
      Wd[(wf + 0) * LDK + k] = f2bf(v.x); Wd[(wf + 1) * LDK + k] = f2bf(v.y);
      Wd[(wf + 2) * LDK + k] = f2bf(v.z); Wd[(wf + 3) * LDK + k] = f2bf(v.w);
    }
#ifdef HAVE_ASYNC_LDS
    wait_async0();  // own async transfers complete before barrier
#endif
    __syncthreads();

    const v16bf a = frag_a(As, wave * 16 + (lane & 15), lane);
#pragma unroll
    for (int nt = 0; nt < 8; ++nt) {
      const v16bf b = frag_b(Wd, nt * 16 + (lane & 15), lane);
      acc[nt] = wmma_bf16(a, b, acc[nt]);
    }
  }

  const int hi8 = (lane & 16) ? 8 : 0;
#pragma unroll
  for (int nt = 0; nt < 8; ++nt) {
    const int hCol = h0 + nt * 16 + (lane & 15);
    float br[R_];
#pragma unroll
    for (int r = 0; r < R_; ++r) br[r] = Bdn[(size_t)hCol * R_ + r];
#pragma unroll
    for (int v = 0; v < 8; ++v) {
      const int t = m0 + wave * 16 + v + hi8;
      const size_t row = (size_t)e * T_ + t;
      const float* sp = sdn + row * R_;
      float o = acc[nt][v];
#pragma unroll
      for (int r = 0; r < R_; ++r) o += sp[r] * br[r];
      out[row * H_ + hCol] = o;
    }
  }
}

// ---------------------------------------------------------------------------
extern "C" void kernel_launch(void* const* d_in, const int* in_sizes, int n_in,
                              void* d_out, int out_size, void* d_ws, size_t ws_size,
                              hipStream_t stream) {
  (void)in_sizes; (void)n_in; (void)out_size; (void)ws_size;
  const float* x   = (const float*)d_in[0];  // hidden_states [8192, 1024]
  const float* wgu = (const float*)d_in[1];  // [8, 1024, 4096]
  const float* wdn = (const float*)d_in[2];  // [8, 2048, 1024]
  const float* Agu = (const float*)d_in[3];  // [8, 1024]
  const float* Bgu = (const float*)d_in[4];  // [4096, 8]
  const float* Adn = (const float*)d_in[5];  // [8, 2048]
  const float* Bdn = (const float*)d_in[6];  // [1024, 8]
  float* out = (float*)d_out;                // [8192, 1024]

  // Workspace layout (~34 MB): s_gu | s_dn | hidden(bf16)
  char* ws = (char*)d_ws;
  float* sgu = (float*)ws;                                  // 8192*8*4   = 256 KB
  float* sdn = (float*)(ws + (size_t)8192 * R_ * 4);        // 256 KB
  unsigned short* hidden =
      (unsigned short*)(ws + (size_t)2 * 8192 * R_ * 4);    // 8192*2048*2 = 32 MB

  lora_s_gu_kernel<<<dim3(1024), dim3(256), 0, stream>>>(x, Agu, sgu);
  gemm_gu_swiglu_kernel<<<dim3(T_ / 128, I_ / 64, E_), dim3(256), 0, stream>>>(
      x, wgu, Bgu, sgu, hidden);
  lora_s_dn_kernel<<<dim3(1024), dim3(256), 0, stream>>>(hidden, Adn, sdn);
  gemm_down_kernel<<<dim3(T_ / 128, H_ / 128, E_), dim3(256), 0, stream>>>(
      hidden, wdn, Bdn, sdn, out);
}
